// MultiHeadAttention_45724221833894
// MI455X (gfx1250) — matrix operand order repaired and checked by reading, not compiled, not run on hardware
//
#include <hip/hip_runtime.h>
#include <hip/hip_bf16.h>

// ---------------------------------------------------------------------------
// Multi-head attention forward for MI455X (gfx1250, wave32, WMMA bf16).
// Pipeline:
//   gemm512<0> x2 : Q,K projections  fp32 -> bf16 head-split [B,H,S,64]
//   gemm512<2>    : V projection     fp32 -> bf16 TRANSPOSED [B,H,64,S]
//   attn_kernel   : flash attention, barrier-free; 32 q-rows per wave so each
//                   K/V fragment feeds two WMMA tiles; QK^T and P·V fragments
//                   loaded straight from global (L2-resident), row-sums via
//                   an all-ones WMMA, only row-max uses the lane butterfly
//   gemm512<1>    : output projection -> fp32 d_out
// ---------------------------------------------------------------------------

typedef __attribute__((ext_vector_type(16))) __bf16 v16bf;
typedef __attribute__((ext_vector_type(8)))  __bf16 bf16x8;
typedef __attribute__((ext_vector_type(8)))  float  v8f;
typedef __attribute__((ext_vector_type(4)))  float  f32x4;

union F8   { v8f v;    float  e[8];  };
union H8   { bf16x8 v; __bf16 e[8];  };
union V16  { v16bf v;  bf16x8 h[2];  };
union V16E { v16bf v;  __bf16 e[16]; };

// A-matrix fragment (16x32 bf16, MxK). Lane l: m = l%16, g = l/16.
// Element i: K = g*8 + i (i<8), K = 16 + g*8 + (i-8) (i>=8).
// Source row-major [m][K], element stride `stride`.
static __device__ __forceinline__ v16bf frag_a(const __bf16* base, int m, int g, int stride) {
    V16 u;
    const __bf16* p = base + m * stride;
    u.h[0] = *reinterpret_cast<const bf16x8*>(p + g * 8);
    u.h[1] = *reinterpret_cast<const bf16x8*>(p + 16 + g * 8);
    return u.v;
}

// B-matrix fragment (32x16 bf16, KxN). Lane l: n = l%16, g = l/16.
// Element i: K = g*8 + i (i<8), K = 16 + g*8 + (i-8) (i>=8). Source laid out [n][K] (K contiguous per n).
static __device__ __forceinline__ v16bf frag_b(const __bf16* base, int n, int g, int stride) {
    V16 u;
    const __bf16* p = base + (size_t)n * stride;
    u.h[0] = *reinterpret_cast<const bf16x8*>(p + g * 8);
    u.h[1] = *reinterpret_cast<const bf16x8*>(p + 16 + g * 8);
    return u.v;
}

static __device__ __forceinline__ v8f wmma_bf16(v16bf a, v16bf b, v8f c) {
    return __builtin_amdgcn_wmma_f32_16x16x32_bf16(false, a, false, b, (short)0, c, false, false);
}

// ---------------------------------------------------------------------------
// GEMM: Y[m][n] = (sum_k X[m][k] * W[n][k] + bias[n]) * scale
// M = 4096 (B*S), N = K = 512. Block 256 threads (8 waves), tile 128x64.
// MODE 0: bf16 head-split [B,H,S,64]      (Q, K)
// MODE 1: fp32 row-major [M,512]          (final output)
// MODE 2: bf16 transposed head-split [B,H,64,S], vectorized stores (V)
// ---------------------------------------------------------------------------
template <int MODE>
__global__ __launch_bounds__(256) void gemm512(const float* __restrict__ X,
                                               const float* __restrict__ W,
                                               const float* __restrict__ bias,
                                               __bf16* __restrict__ outh,
                                               float* __restrict__ outf,
                                               float scale) {
    __shared__ __align__(16) __bf16 ldsA[128 * 32];  // [m][k]
    __shared__ __align__(16) __bf16 ldsB[64 * 32];   // [n][k] (W accessed transposed)

    const int t    = threadIdx.x;
    const int lane = t & 31;
    const int w    = t >> 5;
    const int g    = lane >> 4;
    const int ln   = lane & 15;
    const int m0   = blockIdx.y * 128;
    const int n0   = blockIdx.x * 64;

    F8 acc[4];
#pragma unroll
    for (int j = 0; j < 4; ++j)
#pragma unroll
        for (int r = 0; r < 8; ++r) acc[j].e[r] = 0.f;

    for (int kt = 0; kt < 16; ++kt) {
        const int k0 = kt * 32;
        __syncthreads();
        {   // stage A tile: 128x32 fp32 -> bf16, 16 elements per thread
            const int m  = t >> 1;
            const int kc = (t & 1) * 16;
            const float* src = X + (size_t)(m0 + m) * 512 + k0 + kc;
            f32x4 a0 = *reinterpret_cast<const f32x4*>(src);
            f32x4 a1 = *reinterpret_cast<const f32x4*>(src + 4);
            f32x4 a2 = *reinterpret_cast<const f32x4*>(src + 8);
            f32x4 a3 = *reinterpret_cast<const f32x4*>(src + 12);
            H8 o0, o1;
#pragma unroll
            for (int i = 0; i < 4; ++i) {
                o0.e[i]     = (__bf16)a0[i];
                o0.e[4 + i] = (__bf16)a1[i];
                o1.e[i]     = (__bf16)a2[i];
                o1.e[4 + i] = (__bf16)a3[i];
            }
            *reinterpret_cast<bf16x8*>(&ldsA[m * 32 + kc])     = o0.v;
            *reinterpret_cast<bf16x8*>(&ldsA[m * 32 + kc + 8]) = o1.v;
        }
        {   // stage B tile: W[n0..n0+64)[k0..k0+32) -> [n][k], 8 elements per thread
            const int n  = t >> 2;
            const int kc = (t & 3) * 8;
            const float* src = W + (size_t)(n0 + n) * 512 + k0 + kc;
            f32x4 b0 = *reinterpret_cast<const f32x4*>(src);
            f32x4 b1 = *reinterpret_cast<const f32x4*>(src + 4);
            H8 o;
#pragma unroll
            for (int i = 0; i < 4; ++i) {
                o.e[i]     = (__bf16)b0[i];
                o.e[4 + i] = (__bf16)b1[i];
            }
            *reinterpret_cast<bf16x8*>(&ldsB[n * 32 + kc]) = o.v;
        }
        __syncthreads();

        const v16bf a = frag_a(&ldsA[w * 16 * 32], ln, g, 32);
#pragma unroll
        for (int j = 0; j < 4; ++j) {
            const v16bf b = frag_b(ldsB, j * 16 + ln, g, 32);
            acc[j].v = wmma_bf16(a, b, acc[j].v);
        }
    }

    // Epilogue. C layout: lane l, reg r -> m = 8*(l/16)+r, n = l%16.
#pragma unroll
    for (int j = 0; j < 4; ++j) {
        const int n  = n0 + j * 16 + ln;
        const float bs = bias[n];
        if (MODE == 2) {
            // [B,H,64,S]: 8 consecutive s per lane -> one 16B store
            const int mbase = m0 + w * 16 + g * 8;      // 8-aligned
            const int bb = mbase >> 11, sbase = mbase & 2047;
            const int hh = n >> 6, d = n & 63;
            H8 pack;
#pragma unroll
            for (int r = 0; r < 8; ++r)
                pack.e[r] = (__bf16)((acc[j].e[r] + bs) * scale);
            *reinterpret_cast<bf16x8*>(
                outh + ((size_t)(bb * 8 + hh) * 64 + d) * 2048 + sbase) = pack.v;
        } else {
#pragma unroll
            for (int r = 0; r < 8; ++r) {
                const int m = m0 + w * 16 + g * 8 + r;
                const float val = (acc[j].e[r] + bs) * scale;
                if (MODE == 0) {
                    const int bb = m >> 11, s = m & 2047;
                    const int hh = n >> 6,  d = n & 63;
                    outh[((size_t)(bb * 8 + hh) * 2048 + s) * 64 + d] = (__bf16)val;
                } else {
                    outf[(size_t)m * 512 + n] = val;
                }
            }
        }
    }
}

// ---------------------------------------------------------------------------
// Flash attention, barrier-free. Grid (S/128, B*H), block 128 (4 waves).
// Each wave owns 32 q-rows (two 16-row WMMA tiles) so every K/V fragment is
// reused twice. Q carries 1/sqrt(hd); K in [B,H,S,64]; V in [B,H,64,S].
// Key-padding mask sets scores to -1e6 (reference semantics). Row sums via
// all-ones WMMA; only row-max uses the 16-lane butterfly.
// ---------------------------------------------------------------------------
__global__ __launch_bounds__(128) void attn_kernel(const __bf16* __restrict__ Q,
                                                   const __bf16* __restrict__ K,
                                                   const __bf16* __restrict__ VT,
                                                   const int* __restrict__ valid_lens,
                                                   float* __restrict__ O) {
    __shared__ __align__(16) __bf16 ldsP[4][32 * 64];  // per-wave P tile [qrow][key]

    const int t    = threadIdx.x;
    const int lane = t & 31;
    const int w    = t >> 5;
    const int g    = lane >> 4;
    const int ln   = lane & 15;
    const int bh   = blockIdx.y;               // b*8 + h
    const int b    = bh >> 3;
    const int h    = bh & 7;
    const int q0   = blockIdx.x * 128 + w * 32;
    const int valid = valid_lens[b];

    const __bf16* Qbase  = Q  + ((size_t)bh * 2048 + q0) * 64;
    const __bf16* Kbase  = K  + (size_t)bh * 2048 * 64;
    const __bf16* VTbase = VT + (size_t)bh * 64 * 2048;

    // Q fragments: 2 m-tiles x 2 k-slices of hd=64
    v16bf qa[2][2];
#pragma unroll
    for (int mt = 0; mt < 2; ++mt) {
        qa[mt][0] = frag_a(Qbase + mt * 16 * 64, ln, g, 64);
        qa[mt][1] = frag_a(Qbase + mt * 16 * 64 + 32, ln, g, 64);
    }

    // all-ones B fragment: D = P @ ones gives row sums in C layout
    V16E onesu;
#pragma unroll
    for (int i = 0; i < 16; ++i) onesu.e[i] = (__bf16)1.0f;
    const v16bf ones = onesu.v;

    F8 o[2][4];
    float mrow[2][8], lrow[2][8];
#pragma unroll
    for (int mt = 0; mt < 2; ++mt) {
#pragma unroll
        for (int j = 0; j < 4; ++j)
#pragma unroll
            for (int r = 0; r < 8; ++r) o[mt][j].e[r] = 0.f;
#pragma unroll
        for (int r = 0; r < 8; ++r) { mrow[mt][r] = -3.0e38f; lrow[mt][r] = 0.f; }
    }

    // keys >= valid give exp(-1e6 - m) == 0 in fp32 whenever any unmasked key
    // exists, so skipping fully-masked chunks is exact. valid==0 -> full loop
    // (uniform softmax, matching the reference).
    const int kend = (valid > 0) ? ((valid + 63) & ~63) : 2048;
    __bf16* Pw = ldsP[w];

    for (int k0 = 0; k0 < kend; k0 += 64) {
        // prefetch next chunk's K/V lines to overlap with the softmax phase
        if (k0 + 64 < kend) {
            __builtin_prefetch(Kbase + (size_t)(k0 + 64) * 64, 0, 1);
            __builtin_prefetch(VTbase + (k0 + 64), 0, 1);
        }

        // ---- S = Q @ K^T : 2 m-tiles x 4 n-tiles over 64 keys -------------
        F8 s[2][4];
#pragma unroll
        for (int nt = 0; nt < 4; ++nt) {
            const __bf16* kb = Kbase + (size_t)(k0 + nt * 16) * 64;
            const v16bf b0 = frag_b(kb, ln, g, 64);        // dims 0..31
            const v16bf b1 = frag_b(kb + 32, ln, g, 64);   // dims 32..63
#pragma unroll
            for (int mt = 0; mt < 2; ++mt) {
#pragma unroll
                for (int r = 0; r < 8; ++r) s[mt][nt].e[r] = 0.f;
                s[mt][nt].v = wmma_bf16(qa[mt][0], b0, s[mt][nt].v);
                s[mt][nt].v = wmma_bf16(qa[mt][1], b1, s[mt][nt].v);
            }
        }

        // ---- key-padding mask --------------------------------------------
#pragma unroll
        for (int nt = 0; nt < 4; ++nt) {
            const int key = k0 + nt * 16 + ln;
            if (key >= valid) {
#pragma unroll
                for (int mt = 0; mt < 2; ++mt)
#pragma unroll
                    for (int r = 0; r < 8; ++r) s[mt][nt].e[r] = -1.0e6f;
            }
        }

        // ---- online softmax: max via butterfly, P -> LDS ------------------
#pragma unroll
        for (int mt = 0; mt < 2; ++mt) {
#pragma unroll
            for (int r = 0; r < 8; ++r) {
                float mx = fmaxf(fmaxf(s[mt][0].e[r], s[mt][1].e[r]),
                                 fmaxf(s[mt][2].e[r], s[mt][3].e[r]));
#pragma unroll
                for (int off = 1; off < 16; off <<= 1)
                    mx = fmaxf(mx, __shfl_xor(mx, off, 32));
                const float mnew = fmaxf(mrow[mt][r], mx);
                const float corr = __expf(mrow[mt][r] - mnew);
                mrow[mt][r] = mnew;
                lrow[mt][r] *= corr;
#pragma unroll
                for (int j = 0; j < 4; ++j) o[mt][j].e[r] *= corr;
                const int row = mt * 16 + g * 8 + r;
#pragma unroll
                for (int nt = 0; nt < 4; ++nt) {
                    const float p = __expf(s[mt][nt].e[r] - mnew);
                    Pw[row * 64 + nt * 16 + ln] = (__bf16)p;
                }
            }
        }
        // wave-private LDS round trip: wait for our own ds stores
        asm volatile("s_wait_dscnt 0" ::: "memory");

        v16bf pa[2][2];
#pragma unroll
        for (int mt = 0; mt < 2; ++mt) {
            pa[mt][0] = frag_a(Pw + mt * 16 * 64, ln, g, 64);       // keys +0..31
            pa[mt][1] = frag_a(Pw + mt * 16 * 64 + 32, ln, g, 64);  // keys +32..63
        }

        // ---- row sums via matrix pipe ------------------------------------
#pragma unroll
        for (int mt = 0; mt < 2; ++mt) {
            F8 ls;
#pragma unroll
            for (int r = 0; r < 8; ++r) ls.e[r] = 0.f;
            ls.v = wmma_bf16(pa[mt][0], ones, ls.v);
            ls.v = wmma_bf16(pa[mt][1], ones, ls.v);
#pragma unroll
            for (int r = 0; r < 8; ++r) lrow[mt][r] += ls.e[r];
        }

        // ---- O += P @ V (B fragments straight from transposed global V) ---
#pragma unroll
        for (int j = 0; j < 4; ++j) {
            const v16bf vb0 = frag_b(VTbase + k0,      j * 16 + ln, g, 2048);
            const v16bf vb1 = frag_b(VTbase + k0 + 32, j * 16 + ln, g, 2048);
#pragma unroll
            for (int mt = 0; mt < 2; ++mt) {
                o[mt][j].v = wmma_bf16(pa[mt][0], vb0, o[mt][j].v);
                o[mt][j].v = wmma_bf16(pa[mt][1], vb1, o[mt][j].v);
            }
        }
    }

    // ---- normalize, write merged-head fp32 [B, S, 512] --------------------
#pragma unroll
    for (int mt = 0; mt < 2; ++mt) {
#pragma unroll
        for (int j = 0; j < 4; ++j) {
#pragma unroll
            for (int r = 0; r < 8; ++r) {
                const int row = q0 + mt * 16 + g * 8 + r;
                const float val = o[mt][j].e[r] / lrow[mt][r];
                O[((size_t)b * 2048 + row) * 512 + h * 64 + j * 16 + ln] = val;
            }
        }
    }
}

// ---------------------------------------------------------------------------
extern "C" void kernel_launch(void* const* d_in, const int* in_sizes, int n_in,
                              void* d_out, int out_size, void* d_ws, size_t ws_size,
                              hipStream_t stream) {
    (void)in_sizes; (void)n_in; (void)out_size; (void)ws_size;
    const float* queries = (const float*)d_in[0];
    const float* keys    = (const float*)d_in[1];
    const float* values  = (const float*)d_in[2];
    const int*   vlen    = (const int*)d_in[3];
    const float* Wq = (const float*)d_in[4];
    const float* bq = (const float*)d_in[5];
    const float* Wk = (const float*)d_in[6];
    const float* bk = (const float*)d_in[7];
    const float* Wv = (const float*)d_in[8];
    const float* bv = (const float*)d_in[9];
    const float* Wo = (const float*)d_in[10];
    const float* bo = (const float*)d_in[11];

    // workspace: Q,K bf16 [B,H,S,64], V bf16 [B,H,64,S] (4 MB each),
    // attention output fp32 [B,S,512] (8 MB)
    const size_t QKV_ELEMS = (size_t)2 * 8 * 2048 * 64;
    __bf16* Qb  = (__bf16*)d_ws;
    __bf16* Kb  = Qb + QKV_ELEMS;
    __bf16* VTb = Kb + QKV_ELEMS;
    float*  Ao  = (float*)(VTb + QKV_ELEMS);

    const dim3 ggrid(8, 32);   // N/64 x M/128 (M = 4096, N = 512)
    const dim3 gblk(256);
    // 1/sqrt(64) folded into Q projection
    gemm512<0><<<ggrid, gblk, 0, stream>>>(queries, Wq, bq, Qb,  (float*)nullptr, 0.125f);
    gemm512<0><<<ggrid, gblk, 0, stream>>>(keys,    Wk, bk, Kb,  (float*)nullptr, 1.0f);
    gemm512<2><<<ggrid, gblk, 0, stream>>>(values,  Wv, bv, VTb, (float*)nullptr, 1.0f);

    attn_kernel<<<dim3(2048 / 128, 16), 128, 0, stream>>>(Qb, Kb, VTb, vlen, Ao);

    gemm512<1><<<ggrid, gblk, 0, stream>>>(Ao, Wo, bo, (__bf16*)nullptr, (float*)d_out, 1.0f);
}
